// Model_81844896793224
// MI455X (gfx1250) — compile-verified, matrix operand order checked by reading, not run on hardware
//
#include <hip/hip_runtime.h>
#include <hip/hip_bf16.h>
#include <math.h>

typedef __attribute__((ext_vector_type(16))) _Float16 v16h;
typedef __attribute__((ext_vector_type(8)))  _Float16 v8h;
typedef __attribute__((ext_vector_type(8)))  float    v8f;
typedef _Float16 f16;

#define DM    512
#define NH    8
#define HDIM  64
#define DFF   2048
#define SL    512
#define BB    16
#define MROWS (BB*SL)      // 8192
#define BHN   (BB*NH)      // 128
#define PREDL 336
#define CIN   7

// ---------------- elementwise f32 -> f16 (vectorized x8; n must be multiple of 8) ----------------
__global__ void k_cvt_f16(const float* __restrict__ s, f16* __restrict__ d, long long n){
  long long n8 = n >> 3;
  long long i  = (long long)blockIdx.x*blockDim.x + threadIdx.x;
  long long st = (long long)gridDim.x*blockDim.x;
  for (; i < n8; i += st){
    const float* sp = s + i*8;
    v8h v;
    #pragma unroll
    for (int j = 0; j < 8; j++) v[j] = (f16)sp[j];
    *(v8h*)(d + i*8) = v;
  }
}

// ---------------- im2col for causal conv k=4 (left pad 3), vectorized x8 ----------------
// x [B,512,512] f32 -> out [B*512, 4*512] f16 ; out[m, t*512+d] = x[b, l-3+t, d]
__global__ void k_im2col4(const float* __restrict__ x, f16* __restrict__ out){
  const long long n8 = (long long)MROWS * 4 * DM / 8;
  long long i  = (long long)blockIdx.x*blockDim.x + threadIdx.x;
  long long st = (long long)gridDim.x*blockDim.x;
  for (; i < n8; i += st){
    int g = (int)(i & 255);            // 2048/8 groups per output row
    long long m = i >> 8;
    int col = g * 8;
    int t = col >> 9, d = col & (DM-1);
    int b = (int)(m >> 9), l = (int)(m & (SL-1));
    int ls = l - 3 + t;
    v8h v = {};
    if (ls >= 0){
      const float* src = &x[((long long)(b << 9) + ls) * DM + d];
      #pragma unroll
      for (int j = 0; j < 8; j++) v[j] = (f16)src[j];
    }
    *(v8h*)(out + i*8) = v;
  }
}

// ---------------- embedding: circular token conv k=3 + sinusoid pos + mark@temp ----------------
__global__ void k_embed(const float* __restrict__ xin, int cin,
                        const float* __restrict__ mark,
                        const float* __restrict__ Wtok,   // [3,cin,512]
                        const float* __restrict__ Wtemp,  // [4,512]
                        float* __restrict__ out, f16* __restrict__ out16){
  int row = blockIdx.x;            // b*512 + l
  int b = row >> 9, l = row & (SL-1);
  for (int d = threadIdx.x; d < DM; d += blockDim.x){
    float s = 0.f;
    for (int t = 0; t < 3; t++){
      int l2 = (l - 1 + t + SL) & (SL-1);              // circular pad
      const float* xr = xin  + ((long long)(b << 9) + l2) * cin;
      const float* wr = Wtok + (long long)t * cin * DM + d;
      for (int c = 0; c < cin; c++) s += xr[c] * wr[(long long)c * DM];
    }
    int j = d >> 1;
    float freq = __expf((float)(2*j) * (-9.210340371976184f / (float)DM));
    float ang  = (float)l * freq;
    s += (d & 1) ? __cosf(ang) : __sinf(ang);
    const float* mr = mark + (long long)row * 4;
    for (int m2 = 0; m2 < 4; m2++) s += mr[m2] * Wtemp[m2*DM + d];
    out  [(long long)row*DM + d] = s;
    out16[(long long)row*DM + d] = (f16)s;
  }
}

// ---------------- generic tiled WMMA GEMM ----------------
// C/C16 = act(alpha * A[MxK] @ op(B) + bias), op(B)=B[KxN] (NN) or B[NxK]^T (NT)
// Block tile 128x64x32; 8 waves, each wave computes a 32x32 sub-tile (4 WMMAs/K-step).
// batched: per blockIdx.z, operand offset = (z/batchH)*Outer + (z%batchH)*Inner
#define BM 128
#define BN 64
#define BKK 32
#define LDH 56   // 112B row pitch: 16B-aligned b128 slots, conflict-free 28-bank row step

__global__ __launch_bounds__(256) void k_gemm_wmma(
    const f16* __restrict__ A, const f16* __restrict__ Bm, const float* __restrict__ bias,
    float* __restrict__ C, f16* __restrict__ C16,
    int M, int N, int K, int lda, int ldb, int ldc,
    int transB, float alpha, int act, int batchH,
    long long aO, long long aI, long long bO, long long bI, long long cO, long long cI){
  __shared__ f16 As[BM][LDH];            // 14.0 KB
  __shared__ f16 Bs[BN][LDH];            // 7.0 KB, stored transposed: Bs[n][k]
  const int bz = blockIdx.z;
  A  += (long long)(bz / batchH) * aO + (long long)(bz % batchH) * aI;
  Bm += (long long)(bz / batchH) * bO + (long long)(bz % batchH) * bI;
  const long long coff = (long long)(bz / batchH) * cO + (long long)(bz % batchH) * cI;

  const int m0 = blockIdx.y * BM, n0 = blockIdx.x * BN;
  const int tid = threadIdx.x, wave = tid >> 5, lane = tid & 31;
  const int wm = (wave & 3) * 32;        // 4 wave-groups down 128 rows, 32 rows each
  const int wn = (wave >> 2) * 32;       // 2 wave-groups across 64 cols, 32 cols each
  const int lr = lane & 15, hi = lane >> 4;
  const int kb = hi * 8;                 // ISA 16-bit operand lane layout

  const bool aRows = (m0 + BM <= M);
  const bool bCols = (n0 + BN <= N);

  v8f acc00 = {}, acc01 = {}, acc10 = {}, acc11 = {};

  for (int k0 = 0; k0 < K; k0 += BKK){
    const bool kFull = (k0 + BKK <= K);
    // ---- A tile: 128 rows x 32 k (each thread: 2 rows, 8 halves each) ----
    {
      int r = tid >> 2, c = (tid & 3) * 8;
      #pragma unroll
      for (int rr = 0; rr < 2; rr++){
        int row = r + rr * 64;
        int gm = m0 + row;
        if (aRows && kFull){
          const f16* gp = A + (long long)gm * lda + k0 + c;
          *(v8h*)(&As[row][c]) = *(const v8h*)gp;
          if (k0 + BKK < K) __builtin_prefetch((const void*)(gp + BKK), 0, 3);
        } else {
          #pragma unroll
          for (int i = 0; i < 8; i++){
            int gk = k0 + c + i;
            f16 v = (f16)0.f;
            if (gm < M && gk < K) v = A[(long long)gm * lda + gk];
            As[row][c + i] = v;
          }
        }
      }
    }
    // ---- B tile -> Bs[n][k] ----
    if (transB){
      int nn = tid >> 2, kblk = (tid & 3) * 8;
      int gn = n0 + nn;
      if (bCols && kFull){
        const f16* gp = Bm + (long long)gn * ldb + k0 + kblk;
        *(v8h*)(&Bs[nn][kblk]) = *(const v8h*)gp;
        if (k0 + BKK < K) __builtin_prefetch((const void*)(gp + BKK), 0, 3);
      } else {
        #pragma unroll
        for (int i = 0; i < 8; i++){
          int gk = k0 + kblk + i;
          f16 v = (f16)0.f;
          if (gn < N && gk < K) v = Bm[(long long)gn * ldb + gk];
          Bs[nn][kblk + i] = v;
        }
      }
    } else {
      int kk = tid >> 3, nblk = (tid & 7) * 8;
      int gk = k0 + kk;
      if (bCols && kFull){
        const f16* gp = Bm + (long long)gk * ldb + n0 + nblk;
        v8h bv = *(const v8h*)gp;
        if (k0 + BKK < K) __builtin_prefetch((const void*)(gp + (long long)BKK * ldb), 0, 3);
        #pragma unroll
        for (int i = 0; i < 8; i++) Bs[nblk + i][kk] = bv[i];
      } else {
        #pragma unroll
        for (int i = 0; i < 8; i++){
          int gn = n0 + nblk + i;
          f16 v = (f16)0.f;
          if (gk < K && gn < N) v = Bm[(long long)gk * ldb + gn];
          Bs[nblk + i][kk] = v;
        }
      }
    }
    __syncthreads();
    v16h af0, af1, bf0, bf1;
    {
      const f16* p0 = &As[wm + lr][0];
      #pragma unroll
      for (int i = 0; i < 8; i++){ af0[i] = p0[kb + i]; af0[8 + i] = p0[kb + 16 + i]; }
      const f16* p1 = &As[wm + 16 + lr][0];
      #pragma unroll
      for (int i = 0; i < 8; i++){ af1[i] = p1[kb + i]; af1[8 + i] = p1[kb + 16 + i]; }
      const f16* q0 = &Bs[wn + lr][0];
      #pragma unroll
      for (int i = 0; i < 8; i++){ bf0[i] = q0[kb + i]; bf0[8 + i] = q0[kb + 16 + i]; }
      const f16* q1 = &Bs[wn + 16 + lr][0];
      #pragma unroll
      for (int i = 0; i < 8; i++){ bf1[i] = q1[kb + i]; bf1[8 + i] = q1[kb + 16 + i]; }
    }
    acc00 = __builtin_amdgcn_wmma_f32_16x16x32_f16(false, af0, false, bf0, (short)0, acc00, false, false);
    acc01 = __builtin_amdgcn_wmma_f32_16x16x32_f16(false, af0, false, bf1, (short)0, acc01, false, false);
    acc10 = __builtin_amdgcn_wmma_f32_16x16x32_f16(false, af1, false, bf0, (short)0, acc10, false, false);
    acc11 = __builtin_amdgcn_wmma_f32_16x16x32_f16(false, af1, false, bf1, (short)0, acc11, false, false);
    __syncthreads();
  }

  #pragma unroll
  for (int mt = 0; mt < 2; mt++){
    #pragma unroll
    for (int nt = 0; nt < 2; nt++){
      v8f acc = mt ? (nt ? acc11 : acc10) : (nt ? acc01 : acc00);
      int gn = n0 + wn + nt * 16 + lr;
      #pragma unroll
      for (int r = 0; r < 8; r++){
        int gm = m0 + wm + mt * 16 + hi * 8 + r;   // C layout: lanes16-31 -> M+8
        if (gm < M && gn < N){
          float v = alpha * acc[r] + (bias ? bias[gn] : 0.f);
          if (act == 1) v = 0.5f * v * (1.f + erff(v * 0.7071067811865476f));
          if (C)   C  [coff + (long long)gm * ldc + gn] = v;
          if (C16) C16[coff + (long long)gm * ldc + gn] = (f16)v;
        }
      }
    }
  }
}

// ---------------- log-sparse masked softmax: scores f32 -> probs f16 ----------------
__global__ void k_softmax_logsparse(const float* __restrict__ S, f16* __restrict__ P, int causal){
  int l = blockIdx.x, bh = blockIdx.y, tid = threadIdx.x;
  const float* row  = S + ((long long)bh * SL + l) * SL;
  f16*         prow = P + ((long long)bh * SL + l) * SL;
  __shared__ float red[256];
  float vals[2]; bool ok[2];
  float mx = -1e30f;
  #pragma unroll
  for (int i = 0; i < 2; i++){
    int s = tid + i * 256;
    int d = l - s; if (d < 0) d = -d;
    bool allowed = (d <= 13) || ((d > 0) && ((d & (d - 1)) == 0)) || ((d % 96) == 0);
    if (causal && s > l) allowed = false;
    ok[i] = allowed;
    vals[i] = allowed ? row[s] : -1e30f;
    mx = fmaxf(mx, vals[i]);
  }
  red[tid] = mx; __syncthreads();
  for (int o = 128; o > 0; o >>= 1){ if (tid < o) red[tid] = fmaxf(red[tid], red[tid+o]); __syncthreads(); }
  mx = red[0]; __syncthreads();
  float sum = 0.f;
  #pragma unroll
  for (int i = 0; i < 2; i++){ vals[i] = ok[i] ? __expf(vals[i] - mx) : 0.f; sum += vals[i]; }
  red[tid] = sum; __syncthreads();
  for (int o = 128; o > 0; o >>= 1){ if (tid < o) red[tid] += red[tid+o]; __syncthreads(); }
  float inv = 1.f / red[0];
  #pragma unroll
  for (int i = 0; i < 2; i++) prow[tid + i*256] = (f16)(vals[i] * inv);
}

// ---------------- LayerNorm (+optional residual), dual f32/f16 output ----------------
__global__ void k_layernorm(const float* __restrict__ x, const float* __restrict__ res,
                            const float* __restrict__ g, const float* __restrict__ b,
                            float* __restrict__ y, f16* __restrict__ y16){
  int row = blockIdx.x, tid = threadIdx.x;
  __shared__ float red[256];
  const float* xr = x + (long long)row * DM;
  float v0 = xr[tid], v1 = xr[tid + 256];
  if (res){ const float* rr = res + (long long)row * DM; v0 += rr[tid]; v1 += rr[tid + 256]; }
  red[tid] = v0 + v1; __syncthreads();
  for (int o = 128; o > 0; o >>= 1){ if (tid < o) red[tid] += red[tid+o]; __syncthreads(); }
  float mean = red[0] * (1.f/512.f); __syncthreads();
  float d0 = v0 - mean, d1 = v1 - mean;
  red[tid] = d0*d0 + d1*d1; __syncthreads();
  for (int o = 128; o > 0; o >>= 1){ if (tid < o) red[tid] += red[tid+o]; __syncthreads(); }
  float rstd = rsqrtf(red[0] * (1.f/512.f) + 1e-5f);
  float o0 = d0 * rstd * g[tid] + b[tid];
  float o1 = d1 * rstd * g[tid + 256] + b[tid + 256];
  y[(long long)row*DM + tid] = o0; y[(long long)row*DM + tid + 256] = o1;
  if (y16){ y16[(long long)row*DM + tid] = (f16)o0; y16[(long long)row*DM + tid + 256] = (f16)o1; }
}

// ---------------- final: slice last 336 rows, proj[..,7] @ final_W[7,1] + b ----------------
__global__ void k_final(const float* __restrict__ proj, const float* __restrict__ fW,
                        const float* __restrict__ fb, float* __restrict__ out){
  int i = blockIdx.x * blockDim.x + threadIdx.x;
  if (i >= BB * PREDL) return;
  int b = i / PREDL, p = i % PREDL;
  const float* pr = proj + ((long long)b * SL + (SL - PREDL) + p) * CIN;
  float s = fb[0];
  #pragma unroll
  for (int c = 0; c < CIN; c++) s += pr[c] * fW[c];
  out[i] = s;
}

// ==================== host ====================
struct AttnP { const f16 *Wq,*Wk,*Wv,*Wo; const float *bq,*bk,*bv,*bo; };
struct FfnP  { const f16 *W1,*W2; const float *b1,*b2; };
struct LnP   { const float *g,*b; };

extern "C" void kernel_launch(void* const* d_in, const int* in_sizes, int n_in,
                              void* d_out, int out_size, void* d_ws, size_t ws_size,
                              hipStream_t stream){
  (void)in_sizes; (void)n_in; (void)out_size; (void)ws_size;
  // pytree flatten order: top-level insertion order, params dict keys sorted alphabetically
  enum {
    I_XENC=0, I_XMKE, I_XDEC, I_XMKD,
    I_DCA=4, I_DFFN=12, I_DN1=16, I_DN2=18, I_DN3=20, I_DSA=22,
    I_DNORM=30, I_DTEMP=32, I_DTOK=33,
    I_E0A=34, I_E0F=42, I_E0N1=46, I_E0N2=48,
    I_E1A=50, I_E1F=58, I_E1N1=62, I_E1N2=64,
    I_ENORM=66, I_ETEMP=68, I_ETOK=69,
    I_FW=70, I_FB=71, I_PW=72, I_PB=73
  };
  size_t off = 0;
  auto alloc = [&](size_t bytes)->void*{
    off = (off + 255) & ~(size_t)255; void* p = (char*)d_ws + off; off += bytes; return p;
  };
  auto cvt = [&](int idx, size_t n)->const f16*{
    f16* p = (f16*)alloc(n * sizeof(f16));
    k_cvt_f16<<<256,256,0,stream>>>((const float*)d_in[idx], p, (long long)n);
    return p;
  };
  auto mkattn = [&](int base)->AttnP{           // sorted: Wk,Wo,Wq,Wv,bk,bo,bq,bv
    AttnP a;
    a.Wk = cvt(base+0, (size_t)4*DM*DM); a.Wo = cvt(base+1, (size_t)DM*DM);
    a.Wq = cvt(base+2, (size_t)4*DM*DM); a.Wv = cvt(base+3, (size_t)DM*DM);
    a.bk = (const float*)d_in[base+4]; a.bo = (const float*)d_in[base+5];
    a.bq = (const float*)d_in[base+6]; a.bv = (const float*)d_in[base+7];
    return a;
  };
  auto mkffn = [&](int base)->FfnP{             // sorted: W1,W2,b1,b2
    FfnP f;
    f.W1 = cvt(base+0, (size_t)DM*DFF); f.W2 = cvt(base+1, (size_t)DFF*DM);
    f.b1 = (const float*)d_in[base+2];  f.b2 = (const float*)d_in[base+3];
    return f;
  };
  auto mkln = [&](int base)->LnP{ LnP l; l.b=(const float*)d_in[base]; l.g=(const float*)d_in[base+1]; return l; };

  AttnP encA0 = mkattn(I_E0A), encA1 = mkattn(I_E1A), decSA = mkattn(I_DSA), decCA = mkattn(I_DCA);
  FfnP  encF0 = mkffn(I_E0F),  encF1 = mkffn(I_E1F),  decF  = mkffn(I_DFFN);
  LnP e0n1 = mkln(I_E0N1), e0n2 = mkln(I_E0N2), e1n1 = mkln(I_E1N1), e1n2 = mkln(I_E1N2);
  LnP dn1 = mkln(I_DN1), dn2 = mkln(I_DN2), dn3 = mkln(I_DN3);
  LnP encN = mkln(I_ENORM), decN = mkln(I_DNORM);
  const f16* projW = cvt(I_PW, (size_t)DM*CIN);

  // activation scratch
  float* xe     = (float*)alloc((size_t)MROWS*DM*4);
  f16*   xe16   = (f16*)  alloc((size_t)MROWS*DM*2);
  float* xd     = (float*)alloc((size_t)MROWS*DM*4);
  f16*   xd16   = (f16*)  alloc((size_t)MROWS*DM*2);
  float* encO   = (float*)alloc((size_t)MROWS*DM*4);
  f16*   encO16 = (f16*)  alloc((size_t)MROWS*DM*2);
  f16*   icol   = (f16*)  alloc((size_t)MROWS*4*DM*2);
  f16*   Q16    = (f16*)  alloc((size_t)MROWS*DM*2);
  f16*   K16    = (f16*)  alloc((size_t)MROWS*DM*2);
  f16*   V16    = (f16*)  alloc((size_t)MROWS*DM*2);
  float* S32    = (float*)alloc((size_t)BHN*SL*SL*4);
  f16*   P16    = (f16*)  alloc((size_t)BHN*SL*SL*2);
  f16*   O16    = (f16*)  alloc((size_t)MROWS*DM*2);
  float* attO   = (float*)alloc((size_t)MROWS*DM*4);
  f16*   H16    = (f16*)  alloc((size_t)MROWS*DFF*2);
  float* Ff     = (float*)alloc((size_t)MROWS*DM*4);
  float* PrO    = (float*)alloc((size_t)MROWS*CIN*4);

  auto gemm = [&](const f16* A, const f16* Bw, const float* bias, float* C, f16* C16,
                  int M, int N, int K, int lda, int ldb, int ldc,
                  int transB, float alpha, int act,
                  int batches, int batchH,
                  long long aO, long long aI, long long bO, long long bI,
                  long long cO, long long cI){
    dim3 g((N + BN - 1)/BN, (M + BM - 1)/BM, batches);
    k_gemm_wmma<<<g,256,0,stream>>>(A,Bw,bias,C,C16,M,N,K,lda,ldb,ldc,
                                    transB,alpha,act,batchH,aO,aI,bO,bI,cO,cI);
  };
  const long long PH = (long long)SL*SL;   // 262144

  auto attention = [&](const float* xq32, const float* xkv32, const f16* xkv16,
                       const AttnP& w, int causal, float* outf32){
    k_im2col4<<<2048,256,0,stream>>>(xq32, icol);
    gemm(icol, w.Wq, w.bq, nullptr, Q16, MROWS, DM, 4*DM, 4*DM, DM, DM, 0, 1.f, 0, 1,1,0,0,0,0,0,0);
    k_im2col4<<<2048,256,0,stream>>>(xkv32, icol);
    gemm(icol, w.Wk, w.bk, nullptr, K16, MROWS, DM, 4*DM, 4*DM, DM, DM, 0, 1.f, 0, 1,1,0,0,0,0,0,0);
    gemm(xkv16, w.Wv, w.bv, nullptr, V16, MROWS, DM, DM, DM, DM, DM, 0, 1.f, 0, 1,1,0,0,0,0,0,0);
    // scores[bh] = 0.125 * Q_bh @ K_bh^T   (batched NT over 128 (b,h))
    gemm(Q16, K16, nullptr, S32, nullptr, SL, SL, HDIM, DM, DM, SL, 1, 0.125f, 0,
         BHN, NH, PH, HDIM, PH, HDIM, 8*PH, PH);
    k_softmax_logsparse<<<dim3(SL,BHN),256,0,stream>>>(S32, P16, causal);
    // O[bh] = P_bh @ V_bh                  (batched NN over 128 (b,h))
    gemm(P16, V16, nullptr, nullptr, O16, SL, HDIM, SL, SL, DM, DM, 0, 1.f, 0,
         BHN, NH, 8*PH, PH, PH, HDIM, PH, HDIM);
    gemm(O16, w.Wo, w.bo, outf32, nullptr, MROWS, DM, DM, DM, DM, DM, 0, 1.f, 0, 1,1,0,0,0,0,0,0);
  };
  auto lnorm = [&](const float* x, const float* res, LnP p, float* y, f16* y16){
    k_layernorm<<<MROWS,256,0,stream>>>(x, res, p.g, p.b, y, y16);
  };
  auto ffn = [&](const f16* x16, const FfnP& f){
    gemm(x16, f.W1, f.b1, nullptr, H16, MROWS, DFF, DM, DM, DFF, DFF, 0, 1.f, 1, 1,1,0,0,0,0,0,0);
    gemm(H16, f.W2, f.b2, Ff, nullptr, MROWS, DM, DFF, DFF, DM, DM, 0, 1.f, 0, 1,1,0,0,0,0,0,0);
  };

  // ---- encoder ----
  k_embed<<<MROWS,256,0,stream>>>((const float*)d_in[I_XENC], CIN, (const float*)d_in[I_XMKE],
                                  (const float*)d_in[I_ETOK], (const float*)d_in[I_ETEMP], xe, xe16);
  const AttnP* eA[2] = {&encA0, &encA1}; const FfnP* eF[2] = {&encF0, &encF1};
  LnP eN1[2] = {e0n1, e1n1}, eN2[2] = {e0n2, e1n2};
  for (int li = 0; li < 2; li++){
    attention(xe, xe, xe16, *eA[li], 0, attO);
    lnorm(xe, attO, eN1[li], xe, xe16);       // x = LN1(x + attn)
    ffn(xe16, *eF[li]);
    lnorm(xe, Ff, eN2[li], xe, xe16);         // x = LN2(x + ffn)
  }
  lnorm(xe, nullptr, encN, encO, encO16);     // encoder output

  // ---- decoder ----
  k_embed<<<MROWS,256,0,stream>>>((const float*)d_in[I_XDEC], CIN, (const float*)d_in[I_XMKD],
                                  (const float*)d_in[I_DTOK], (const float*)d_in[I_DTEMP], xd, xd16);
  attention(xd, xd, xd16, decSA, 1, attO);    // causal log-sparse self-attn
  lnorm(xd, attO, dn1, xd, xd16);
  attention(xd, encO, encO16, decCA, 0, attO);// cross-attn vs encoder
  lnorm(xd, attO, dn2, xd, xd16);
  ffn(xd16, decF);
  lnorm(xd, Ff, dn3, xd, xd16);
  lnorm(xd, nullptr, decN, xd, xd16);

  // ---- projections ----
  gemm(xd16, projW, (const float*)d_in[I_PB], PrO, nullptr, MROWS, CIN, DM, DM, CIN, CIN,
       0, 1.f, 0, 1,1,0,0,0,0,0,0);
  k_final<<<(BB*PREDL + 255)/256,256,0,stream>>>(PrO, (const float*)d_in[I_FW],
                                                 (const float*)d_in[I_FB], (float*)d_out);
}
